// ViewAwareSABlock_50843822850697
// MI455X (gfx1250) — matrix operand order repaired and checked by reading, not compiled, not run on hardware
//
#include <hip/hip_runtime.h>
#include <hip/hip_bf16.h>
#include <math.h>

typedef __attribute__((ext_vector_type(16))) _Float16 v16h;
typedef __attribute__((ext_vector_type(8)))  _Float16 v8h;
typedef __attribute__((ext_vector_type(8)))  float    v8f;

#define DEV static __device__ __forceinline__

// ---------------- problem sizes ----------------
#define Csz   256
#define HEADS 8
#define HD    32
#define BF    128      // B * S * S
#define NTOK  256      // tokens per sparse group (16x16)
#define MTOT  32768    // BF*NTOK == B*H*W
#define NTOPK 64
#define MLPD  1024

// ---------------- WMMA fragment helpers (gfx1250, wave32) ----------------
// A fragment: 16(M) x 32(K), 16-bit. lane<16: M=lane, halves 0..7 -> K 0..7,
// halves 8..15 -> K 16..23. lane>=16: M=lane-16, K 8..15 / 24..31.
DEV v16h ld_fragA(const _Float16* A, int lda, int m0, int k0) {
  int lane = threadIdx.x & 31;
  const _Float16* p = A + (size_t)(m0 + (lane & 15)) * lda + k0 + ((lane & 16) ? 8 : 0);
  union { v16h v; v8h h[2]; } u;
  u.h[0] = *(const v8h*)p;
  u.h[1] = *(const v8h*)(p + 16);
  return u.v;
}
// B fragment: 32(K) x 16(N) built from row-major W[N][K]. lane<16: N=n0+lane,
// halves 0..7 -> K 0..7, halves 8..15 -> K 16..23. lane>=16: K 8..15 / 24..31.
DEV v16h ld_fragB(const _Float16* W, int ldw, int n0, int k0) {
  int lane = threadIdx.x & 31;
  const _Float16* p = W + (size_t)(n0 + (lane & 15)) * ldw + k0 + ((lane & 16) ? 8 : 0);
  union { v16h v; v8h h[2]; } u;
  u.h[0] = *(const v8h*)p;
  u.h[1] = *(const v8h*)(p + 16);
  return u.v;
}
DEV v8f wmma16(v16h a, v16h b, v8f c) {
  return __builtin_amdgcn_wmma_f32_16x16x32_f16(false, a, false, b, (short)0, c, false, false);
}
DEV void gemm_core(const _Float16* A, const _Float16* W, int K, int m0, int n0, v8f acc[4]) {
  for (int k0 = 0; k0 < K; k0 += 32) {
    v16h a = ld_fragA(A, K, m0, k0);
#pragma unroll
    for (int j = 0; j < 4; ++j) {
      v16h b = ld_fragB(W, K, n0 + j * 16, k0);
      acc[j] = wmma16(a, b, acc[j]);
    }
  }
}

// ---------------- misc device helpers ----------------
DEV float geluf(float x) { return 0.5f * x * (1.0f + erff(x * 0.70710678118654752f)); }
DEV unsigned sortkey(float f) {            // monotone float -> uint mapping
  unsigned u = __float_as_uint(f);
  return u ^ ((unsigned)((int)u >> 31) | 0x80000000u);
}
// reductions confined to 16-lane halves (xor masks < 16)
DEV int hsum_i(int v) {
  v += __shfl_xor(v, 1, 32); v += __shfl_xor(v, 2, 32);
  v += __shfl_xor(v, 4, 32); v += __shfl_xor(v, 8, 32); return v;
}
DEV float hsum_f(float v) {
  v += __shfl_xor(v, 1, 32); v += __shfl_xor(v, 2, 32);
  v += __shfl_xor(v, 4, 32); v += __shfl_xor(v, 8, 32); return v;
}
DEV float hmax_f(float v) {
  v = fmaxf(v, __shfl_xor(v, 1, 32)); v = fmaxf(v, __shfl_xor(v, 2, 32));
  v = fmaxf(v, __shfl_xor(v, 4, 32)); v = fmaxf(v, __shfl_xor(v, 8, 32)); return v;
}

// ---------------- kernels ----------------
__global__ void k_cvt_f16(const float* __restrict__ s, _Float16* __restrict__ d, int n) {
  int i = blockIdx.x * 256 + threadIdx.x;
  if (i < n) d[i] = (_Float16)s[i];
}

// sparse gather + LayerNorm1.  block = 256 threads = one token.
__global__ void k_sparse_ln(const float* __restrict__ x, const float* __restrict__ nw,
                            const float* __restrict__ nb, float* __restrict__ xt32,
                            _Float16* __restrict__ xt16) {
  int tokid = blockIdx.x;                 // bf*256 + t
  int bf = tokid >> 8, t = tokid & 255;
  int b = bf >> 4, sh = (bf >> 2) & 3, sw = bf & 3;
  int h = (t >> 4) * 4 + sh, w = (t & 15) * 4 + sw;
  int c = threadIdx.x;
  float v = x[(((size_t)(b * 256 + c)) * 64 + h) * 64 + w];
  __shared__ float red[256];
  red[c] = v; __syncthreads();
  for (int s2 = 128; s2 > 0; s2 >>= 1) { if (c < s2) red[c] += red[c + s2]; __syncthreads(); }
  float mean = red[0] * (1.0f / 256.0f);
  __syncthreads();
  float d = v - mean;
  red[c] = d * d; __syncthreads();
  for (int s2 = 128; s2 > 0; s2 >>= 1) { if (c < s2) red[c] += red[c + s2]; __syncthreads(); }
  float rstd = rsqrtf(red[0] * (1.0f / 256.0f) + 1e-5f);
  float o = d * rstd * nw[c] + nb[c];
  xt32[(size_t)tokid * 256 + c] = o;
  xt16[(size_t)tokid * 256 + c] = (_Float16)o;
}

// LayerNorm2 on y (contiguous rows)
__global__ void k_ln2(const float* __restrict__ y, const float* __restrict__ nw,
                      const float* __restrict__ nb, _Float16* __restrict__ o16) {
  int tok = blockIdx.x; int c = threadIdx.x;
  float v = y[(size_t)tok * 256 + c];
  __shared__ float red[256];
  red[c] = v; __syncthreads();
  for (int s2 = 128; s2 > 0; s2 >>= 1) { if (c < s2) red[c] += red[c + s2]; __syncthreads(); }
  float mean = red[0] * (1.0f / 256.0f);
  __syncthreads();
  float d = v - mean;
  red[c] = d * d; __syncthreads();
  for (int s2 = 128; s2 > 0; s2 >>= 1) { if (c < s2) red[c] += red[c + s2]; __syncthreads(); }
  float rstd = rsqrtf(red[0] * (1.0f / 256.0f) + 1e-5f);
  o16[(size_t)tok * 256 + c] = (_Float16)(d * rstd * nw[c] + nb[c]);
}

// view MLP: LN(vln) -> fc(256->64) gelu -> fc(64->3) softmax -> w_ns
__global__ void k_view_mlp(const float* __restrict__ xt, const float* __restrict__ vlnw,
                           const float* __restrict__ vlnb, const float* __restrict__ w1,
                           const float* __restrict__ b1, const float* __restrict__ w2,
                           const float* __restrict__ b2, float* __restrict__ wns) {
  int tok = blockIdx.x, tid = threadIdx.x;     // 64 threads
  __shared__ float lnv[256]; __shared__ float red[64]; __shared__ float hid[64];
  const float* xr = xt + (size_t)tok * 256;
  float x4[4]; float p = 0.f;
#pragma unroll
  for (int i = 0; i < 4; ++i) { x4[i] = xr[tid * 4 + i]; p += x4[i]; }
  red[tid] = p; __syncthreads();
  for (int s = 32; s > 0; s >>= 1) { if (tid < s) red[tid] += red[tid + s]; __syncthreads(); }
  float mean = red[0] * (1.0f / 256.0f); __syncthreads();
  p = 0.f;
#pragma unroll
  for (int i = 0; i < 4; ++i) { float d = x4[i] - mean; p += d * d; }
  red[tid] = p; __syncthreads();
  for (int s = 32; s > 0; s >>= 1) { if (tid < s) red[tid] += red[tid + s]; __syncthreads(); }
  float rstd = rsqrtf(red[0] * (1.0f / 256.0f) + 1e-5f); __syncthreads();
#pragma unroll
  for (int i = 0; i < 4; ++i) {
    int c = tid * 4 + i;
    lnv[c] = (x4[i] - mean) * rstd * vlnw[c] + vlnb[c];
  }
  __syncthreads();
  float acc = b1[tid];
  const float* wr = w1 + (size_t)tid * 256;
  for (int c = 0; c < 256; ++c) acc += lnv[c] * wr[c];
  hid[tid] = geluf(acc);
  __syncthreads();
  if (tid == 0) {
    float l0 = b2[0], l1 = b2[1], l2 = b2[2];
    for (int i = 0; i < 64; ++i) { float h = hid[i]; l0 += h * w2[i]; l1 += h * w2[64 + i]; l2 += h * w2[128 + i]; }
    float mx = fmaxf(l0, fmaxf(l1, l2));
    float e0 = __expf(l0 - mx), e1 = __expf(l1 - mx), e2 = __expf(l2 - mx);
    wns[tok] = (e1 + e2) / (e0 + e1 + e2);
  }
}

// QKV GEMM: [32768,256]f16 x [768,256]f16^T.  q scaled by d^-1/2, v stored transposed.
__global__ void k_gemm_qkv(const _Float16* __restrict__ A, const _Float16* __restrict__ W,
                           _Float16* __restrict__ qd, _Float16* __restrict__ kd,
                           _Float16* __restrict__ vT) {
  int lane = threadIdx.x & 31, wave = threadIdx.x >> 5;
  int m0 = blockIdx.x * 64 + wave * 16, n0 = blockIdx.y * 64;
  v8f acc[4] = {};
  gemm_core(A, W, 256, m0, n0, acc);
#pragma unroll
  for (int j = 0; j < 4; ++j)
#pragma unroll
    for (int r = 0; r < 8; ++r) {
      int m = m0 + r + ((lane & 16) ? 8 : 0);
      int n = n0 + j * 16 + (lane & 15);
      float v = acc[j][r];
      int which = n >> 8, head = (n >> 5) & 7, d = n & 31;
      int bf = m >> 8, t = m & 255;
      size_t bh = (size_t)(bf * 8 + head);
      if (which == 0)      qd[(bh * 256 + t) * 32 + d] = (_Float16)(v * 0.17677669529663689f);
      else if (which == 1) kd[(bh * 256 + t) * 32 + d] = (_Float16)v;
      else                 vT[(bh * 32 + d) * 256 + t] = (_Float16)v;
    }
}

// attention: 1 block per (group, head); 4 waves, each 4 query tiles of 16.
__global__ void k_attn(const _Float16* __restrict__ qg, const _Float16* __restrict__ kg,
                       const _Float16* __restrict__ vTg, const float* __restrict__ wns,
                       _Float16* __restrict__ aout) {
  int bh = blockIdx.x;
  int bf = bh >> 3, head = bh & 7;
  int lane = threadIdx.x & 31, wave = threadIdx.x >> 5;
  const _Float16* qb = qg + (size_t)bh * 8192;   // [256 tok][32 d]
  const _Float16* kb = kg + (size_t)bh * 8192;   // [256 tok][32 d]
  const _Float16* vb = vTg + (size_t)bh * 8192;  // [32 d][256 tok]
  const float* wrow = wns + bf * 256;
  __shared__ __align__(16) _Float16 plds[4][4096];
  _Float16* pl = plds[wave];

  for (int qt = wave; qt < 16; qt += 4) {
    v16h qa = ld_fragA(qb, 32, qt * 16, 0);
    v8f s[16];
#pragma unroll
    for (int t = 0; t < 16; ++t) {                 // S = Q K^T  (scaled q)
      v16h kf = ld_fragB(kb, 32, t * 16, 0);
      v8f z = {};
      s[t] = wmma16(qa, kf, z);
    }
#pragma unroll
    for (int t = 0; t < 16; ++t) {                 // + ALPHA * w_ns[key]
      float bias = wrow[t * 16 + (lane & 15)];
#pragma unroll
      for (int r = 0; r < 8; ++r) s[t][r] += bias;
    }
    // rows r (lanes 0-15) and r+8 (lanes 16-31) share accumulator index r.
#pragma unroll
    for (int r = 0; r < 8; ++r) {
      float va[16]; unsigned ka[16];
#pragma unroll
      for (int t = 0; t < 16; ++t) { va[t] = s[t][r]; ka[t] = sortkey(va[t]); }
      // radix-select the 64th-largest key over the 256 row values
      unsigned prefix = 0u;
      for (int bit = 31; bit >= 0; --bit) {
        unsigned cand = prefix | (1u << bit);
        int cnt = 0;
#pragma unroll
        for (int t = 0; t < 16; ++t) cnt += (ka[t] >= cand) ? 1 : 0;
        cnt = hsum_i(cnt);
        prefix = (cnt >= NTOPK) ? cand : prefix;
      }
      float mx = -3.0e38f;
#pragma unroll
      for (int t = 0; t < 16; ++t) mx = fmaxf(mx, va[t]);
      mx = hmax_f(mx);
      float pe[16]; float sum = 0.f;
#pragma unroll
      for (int t = 0; t < 16; ++t) {
        float e = (ka[t] >= prefix) ? __expf(va[t] - mx) : 0.f;
        pe[t] = e; sum += e;
      }
      sum = hsum_f(sum);
      float inv = 1.0f / sum;
      int row = r + ((lane & 16) ? 8 : 0);
#pragma unroll
      for (int t = 0; t < 16; ++t)
        pl[row * 256 + t * 16 + (lane & 15)] = (_Float16)(pe[t] * inv);
    }
    asm volatile("s_wait_dscnt 0" ::: "memory");   // cross-lane LDS RAW within wave
    v8f o0 = {}, o1 = {};
#pragma unroll
    for (int k0 = 0; k0 < 256; k0 += 32) {         // O = P V
      v16h pa = ld_fragA(pl, 256, 0, k0);
      v16h v0 = ld_fragB(vb, 256, 0, k0);
      v16h v1 = ld_fragB(vb, 256, 16, k0);
      o0 = wmma16(pa, v0, o0);
      o1 = wmma16(pa, v1, o1);
    }
    int tokbase = bf * 256 + qt * 16;
#pragma unroll
    for (int r = 0; r < 8; ++r) {
      int row = r + ((lane & 16) ? 8 : 0);
      size_t off = (size_t)(tokbase + row) * 256 + head * 32;
      aout[off + (lane & 15)] = (_Float16)o0[r];
      aout[off + 16 + (lane & 15)] = (_Float16)o1[r];
    }
  }
}

// proj GEMM + unsparse + residual:  y = x + gamma1 * (a @ Wp^T + bp)
__global__ void k_gemm_proj(const _Float16* __restrict__ A, const _Float16* __restrict__ W,
                            const float* __restrict__ pb, const float* __restrict__ x,
                            const float* __restrict__ g1, float* __restrict__ y) {
  int lane = threadIdx.x & 31, wave = threadIdx.x >> 5;
  int m0 = blockIdx.x * 64 + wave * 16, n0 = blockIdx.y * 64;
  v8f acc[4] = {};
  gemm_core(A, W, 256, m0, n0, acc);
#pragma unroll
  for (int j = 0; j < 4; ++j)
#pragma unroll
    for (int r = 0; r < 8; ++r) {
      int m = m0 + r + ((lane & 16) ? 8 : 0);
      int n = n0 + j * 16 + (lane & 15);
      float v = acc[j][r] + pb[n];
      int bf = m >> 8, t = m & 255;
      int b = bf >> 4, sh = (bf >> 2) & 3, sw = bf & 3;
      int h = (t >> 4) * 4 + sh, w = (t & 15) * 4 + sw;
      int tok = b * 4096 + h * 64 + w;
      float xv = x[(((size_t)(b * 256 + n)) * 64 + h) * 64 + w];
      y[(size_t)tok * 256 + n] = xv + g1[n] * v;
    }
}

// fc1 GEMM: [32768,256] x [1024,256]^T, +bias -> f16
__global__ void k_gemm_fc1(const _Float16* __restrict__ A, const _Float16* __restrict__ W,
                           const float* __restrict__ b1, _Float16* __restrict__ z1) {
  int lane = threadIdx.x & 31, wave = threadIdx.x >> 5;
  int m0 = blockIdx.x * 64 + wave * 16, n0 = blockIdx.y * 64;
  v8f acc[4] = {};
  gemm_core(A, W, 256, m0, n0, acc);
#pragma unroll
  for (int j = 0; j < 4; ++j)
#pragma unroll
    for (int r = 0; r < 8; ++r) {
      int m = m0 + r + ((lane & 16) ? 8 : 0);
      int n = n0 + j * 16 + (lane & 15);
      z1[(size_t)m * 1024 + n] = (_Float16)(acc[j][r] + b1[n]);
    }
}

// depthwise 3x3 SAME + bias + gelu
__global__ void k_dwconv(const _Float16* __restrict__ z1, const float* __restrict__ dww,
                         const float* __restrict__ dwb, _Float16* __restrict__ z2) {
  int idx = blockIdx.x * 256 + threadIdx.x;       // < 8*4096*1024
  int ch = idx & 1023;
  int pos = idx >> 10;
  int w = pos & 63, h = (pos >> 6) & 63, b = pos >> 12;
  float acc = dwb[ch];
#pragma unroll
  for (int dy = -1; dy <= 1; ++dy)
#pragma unroll
    for (int dx = -1; dx <= 1; ++dx) {
      int hh = h + dy, ww = w + dx;
      if (hh >= 0 && hh < 64 && ww >= 0 && ww < 64)
        acc += (float)z1[((size_t)(b * 4096 + hh * 64 + ww)) * 1024 + ch] *
               dww[((dy + 1) * 3 + (dx + 1)) * 1024 + ch];
    }
  z2[(size_t)idx] = (_Float16)geluf(acc);
}

// fc2 GEMM + residual + NCHW transpose epilogue
__global__ void k_gemm_fc2(const _Float16* __restrict__ A, const _Float16* __restrict__ W,
                           const float* __restrict__ b2, const float* __restrict__ y,
                           const float* __restrict__ g2, float* __restrict__ out) {
  int lane = threadIdx.x & 31, wave = threadIdx.x >> 5;
  int m0 = blockIdx.x * 64 + wave * 16, n0 = blockIdx.y * 64;
  v8f acc[4] = {};
  gemm_core(A, W, 1024, m0, n0, acc);
#pragma unroll
  for (int j = 0; j < 4; ++j)
#pragma unroll
    for (int r = 0; r < 8; ++r) {
      int m = m0 + r + ((lane & 16) ? 8 : 0);
      int n = n0 + j * 16 + (lane & 15);
      float v = acc[j][r] + b2[n];
      int b = m >> 12, hw = m & 4095;
      out[((size_t)(b * 256 + n)) * 4096 + hw] = y[(size_t)m * 256 + n] + g2[n] * v;
    }
}

// ---------------- host ----------------
extern "C" void kernel_launch(void* const* d_in, const int* in_sizes, int n_in,
                              void* d_out, int out_size, void* d_ws, size_t ws_size,
                              hipStream_t stream) {
  (void)in_sizes; (void)n_in; (void)out_size; (void)ws_size;
  const float* x       = (const float*)d_in[0];
  const float* norm1_w = (const float*)d_in[1];
  const float* norm1_b = (const float*)d_in[2];
  const float* vln_w   = (const float*)d_in[3];
  const float* vln_b   = (const float*)d_in[4];
  const float* vfc1_w  = (const float*)d_in[5];
  const float* vfc1_b  = (const float*)d_in[6];
  const float* vfc2_w  = (const float*)d_in[7];
  const float* vfc2_b  = (const float*)d_in[8];
  const float* qkv_w   = (const float*)d_in[9];
  const float* proj_w  = (const float*)d_in[10];
  const float* proj_b  = (const float*)d_in[11];
  const float* norm2_w = (const float*)d_in[12];
  const float* norm2_b = (const float*)d_in[13];
  const float* fc1_w   = (const float*)d_in[14];
  const float* fc1_b   = (const float*)d_in[15];
  const float* dw_w    = (const float*)d_in[16];
  const float* dw_b    = (const float*)d_in[17];
  const float* fc2_w   = (const float*)d_in[18];
  const float* fc2_b   = (const float*)d_in[19];
  const float* gamma1  = (const float*)d_in[20];
  const float* gamma2  = (const float*)d_in[21];
  float* out = (float*)d_out;

  char* ws = (char*)d_ws;
  float*    xt32    = (float*)   (ws + 0);            // 33.6 MB
  _Float16* xt16    = (_Float16*)(ws + 33554432);     // 16.8 MB (reused as ln2 later)
  float*    wns     = (float*)   (ws + 50331648);     // 128 KB
  _Float16* q16     = (_Float16*)(ws + 50462720);     // 16.8 MB
  _Float16* k16     = q16 + (size_t)8388608;
  _Float16* vT16    = k16 + (size_t)8388608;
  _Float16* a16     = (_Float16*)(ws + 100794368);    // 16.8 MB
  float*    y32     = (float*)   (ws + 117571584);    // 33.6 MB
  _Float16* z1      = (_Float16*)(ws + 151126016);    // 67 MB
  _Float16* z2      = (_Float16*)(ws + 218234880);    // 67 MB
  _Float16* qkvw16  = (_Float16*)(ws + 285343744);
  _Float16* projw16 = (_Float16*)(ws + 285736960);
  _Float16* fc1w16  = (_Float16*)(ws + 285868032);
  _Float16* fc2w16  = (_Float16*)(ws + 286392320);
  _Float16* ln216   = xt16;                           // reuse: xt16 dead after qkv gemm

  // weight converts
  k_cvt_f16<<<(768 * 256 + 255) / 256, 256, 0, stream>>>(qkv_w, qkvw16, 768 * 256);
  k_cvt_f16<<<(256 * 256 + 255) / 256, 256, 0, stream>>>(proj_w, projw16, 256 * 256);
  k_cvt_f16<<<(1024 * 256 + 255) / 256, 256, 0, stream>>>(fc1_w, fc1w16, 1024 * 256);
  k_cvt_f16<<<(256 * 1024 + 255) / 256, 256, 0, stream>>>(fc2_w, fc2w16, 256 * 1024);

  k_sparse_ln<<<MTOT, 256, 0, stream>>>(x, norm1_w, norm1_b, xt32, xt16);
  k_view_mlp<<<MTOT, 64, 0, stream>>>(xt32, vln_w, vln_b, vfc1_w, vfc1_b, vfc2_w, vfc2_b, wns);
  k_gemm_qkv<<<dim3(512, 12), 128, 0, stream>>>(xt16, qkvw16, q16, k16, vT16);
  k_attn<<<BF * HEADS, 128, 0, stream>>>(q16, k16, vT16, wns, a16);
  k_gemm_proj<<<dim3(512, 4), 128, 0, stream>>>(a16, projw16, proj_b, x, gamma1, y32);
  k_ln2<<<MTOT, 256, 0, stream>>>(y32, norm2_w, norm2_b, ln216);
  k_gemm_fc1<<<dim3(512, 16), 128, 0, stream>>>(ln216, fc1w16, fc1_b, z1);
  k_dwconv<<<(MTOT * MLPD) / 256, 256, 0, stream>>>(z1, dw_w, dw_b, z2);
  k_gemm_fc2<<<dim3(512, 4), 128, 0, stream>>>(z2, fc2w16, fc2_b, y32, gamma2, out);
}
